// NequIP_39024072851884
// MI455X (gfx1250) — matrix operand order repaired and checked by reading, not compiled, not run on hardware
//
#include <hip/hip_runtime.h>
#include <hip/hip_bf16.h>
#include <math.h>

#define NNODES 20000
#define NEDGES 240000
#define C_S 0.3826834323650898f
#define C_X 0.9238795325112867f

typedef __attribute__((ext_vector_type(16))) __bf16 v16bf;
typedef __attribute__((ext_vector_type(8)))  float  v8f;

union Frag16 { v16bf v; unsigned short h[16]; unsigned int u[8]; uint4 q[2]; };

__device__ __forceinline__ unsigned short f2bf(float f) {
  unsigned int u = __float_as_uint(f);
  u += 0x7FFFu + ((u >> 16) & 1u);          // round-to-nearest-even
  return (unsigned short)(u >> 16);
}
__device__ __forceinline__ float sigm(float x) { return 1.0f / (1.0f + __expf(-x)); }

// ---------------- elementwise / prep kernels ----------------

__global__ void k_prep_sh1(const float* __restrict__ ev, float* __restrict__ sh1, int E) {
  int e = blockIdx.x * blockDim.x + threadIdx.x;
  if (e >= E) return;
  float x = ev[3*e], y = ev[3*e+1], z = ev[3*e+2];
  float inv = 1.7320508075688772f * rsqrtf(x*x + y*y + z*z);   // sqrt(3) * unit vec
  sh1[3*e] = x*inv; sh1[3*e+1] = y*inv; sh1[3*e+2] = z*inv;
}

// h_edge (E x 16 f32) -> padded bf16 (E x 32), cols 16..31 = 0
__global__ void k_cvt_hedge(const float* __restrict__ src, unsigned short* __restrict__ dst,
                            long long total) {
  long long i = (long long)blockIdx.x * blockDim.x + threadIdx.x;
  if (i >= total) return;
  int c = (int)(i & 31);
  long long r = i >> 5;
  dst[i] = (c < 16) ? f2bf(src[r*16 + c]) : (unsigned short)0;
}

// Weight stage: src is (Ksrc x Nc) row-major f32; dst is TRANSPOSED (Nc x Kdst) bf16,
// zero-padded for k in [Ksrc, Kdst). Gives the GEMM contiguous B-fragment loads.
__global__ void k_f2bf_T(const float* __restrict__ src, unsigned short* __restrict__ dst,
                         long long total, int Nc, int Ksrc, int Kdst) {
  long long i = (long long)blockIdx.x * blockDim.x + threadIdx.x;
  if (i >= total) return;                   // total = Nc * Kdst
  int n = (int)(i / Kdst), k = (int)(i % Kdst);
  dst[i] = (k < Ksrc) ? f2bf(src[(size_t)k * Nc + n]) : (unsigned short)0;
}

// A[r, u*8+v] = s[r,u] * z[r % N, v]   (Z == 8 everywhere)
__global__ void k_build_A(const float* __restrict__ s, const float* __restrict__ z,
                          unsigned short* __restrict__ A, long long total, int cols, int ns) {
  long long i = (long long)blockIdx.x * blockDim.x + threadIdx.x;
  if (i >= total) return;
  long long r = i / cols;
  int rem = (int)(i % cols);
  int u = rem >> 3, v = rem & 7;
  int n = (int)(r % NNODES);
  A[i] = f2bf(s[r * (long long)ns + u] * z[n*8 + v]);
}

// bias + silu + bf16, with destination column padding (cols >= Nc written as 0)
__global__ void k_bias_silu_bf(const float* __restrict__ src, const float* __restrict__ bias,
                               unsigned short* __restrict__ dst, long long total,
                               int Nc, int dstStride) {
  long long i = (long long)blockIdx.x * blockDim.x + threadIdx.x;
  if (i >= total) return;                   // total = M * dstStride
  long long r = i / dstStride;
  int c = (int)(i % dstStride);
  if (c < Nc) {
    float t = src[r*Nc + c] + bias[c];
    dst[i] = f2bf(t * sigm(t));
  } else {
    dst[i] = 0;
  }
}

__global__ void k_bias_inplace(float* __restrict__ buf, const float* __restrict__ bias,
                               long long total, int Nc) {
  long long i = (long long)blockIdx.x * blockDim.x + threadIdx.x;
  if (i >= total) return;
  buf[i] += bias[(int)(i % Nc)];
}

// ---------------- WMMA bf16 GEMM: C(MxN) = A(MxK) * Bt^T * scale ----------------
// A: M x K row-major bf16, K % 32 == 0 (zero-padded if needed)
// Bt: N x K row-major bf16 (i.e. B transposed), K % 32 == 0
// One 16x16 C tile per wave32; branch-free inner loop: 4x b128 loads + 1 wmma.

__global__ void __launch_bounds__(256)
k_gemm_bf16(const unsigned short* __restrict__ A, const unsigned short* __restrict__ Bt,
            float* __restrict__ C, int M, int Nc, int K, float scale) {
  int wid  = (int)(((long long)blockIdx.x * blockDim.x + threadIdx.x) >> 5);
  int lane = threadIdx.x & 31;
  int tiles_n = Nc >> 4;
  int tiles_m = M >> 4;
  if (wid >= tiles_m * tiles_n) return;
  int tm = wid / tiles_n, tn = wid % tiles_n;
  int row = lane & 15;
  int hi  = lane >> 4;

  // A frag: lane holds row (tm*16+row); K pairs at k0+hi*8..+7 and +16..+23 (16B each)
  const unsigned short* Arow = A  + (size_t)(tm*16 + row) * K + hi*8;
  // B frag: lane holds col (tn*16+row); K values at k0+hi*8..+7 and +16..+23 (16B each)
  const unsigned short* Brow = Bt + (size_t)(tn*16 + row) * K + hi*8;

  v8f acc = {0.f,0.f,0.f,0.f,0.f,0.f,0.f,0.f};

  for (int k0 = 0; k0 < K; k0 += 32) {
    Frag16 a, b;
    a.q[0] = *(const uint4*)(Arow + k0);
    a.q[1] = *(const uint4*)(Arow + k0 + 16);
    b.q[0] = *(const uint4*)(Brow + k0);
    b.q[1] = *(const uint4*)(Brow + k0 + 16);
    acc = __builtin_amdgcn_wmma_f32_16x16x32_bf16(false, a.v, false, b.v,
                                                  (short)0, acc, false, false);
  }

  float* Crow = C + (size_t)(tm*16) * Nc + tn*16;
#pragma unroll
  for (int g = 0; g < 8; ++g)
    Crow[(size_t)(g + 8*hi) * Nc + (lane & 15)] = acc[g] * scale;
}

// ---------------- message construction + segment-sum (atomic scatter) ----------------

// Layer 0: ns=8, nv=0, nw=16.  agg_s: Nx8, agg_v: [m][n][8]
__global__ void k_msg0(const int* __restrict__ ei, const float* __restrict__ hs,
                       const float* __restrict__ w, const float* __restrict__ sh1,
                       float* __restrict__ agg_s, float* __restrict__ agg_v) {
  long long i = (long long)blockIdx.x * blockDim.x + threadIdx.x;
  if (i >= (long long)NEDGES * 16) return;
  int e = (int)(i >> 4), c = (int)(i & 15);
  int src = ei[e], dst = ei[NEDGES + e];
  const float s12 = 0.28867513459481287f;   // 1/sqrt(12)
  if (c < 8) {
    atomicAdd(&agg_s[(size_t)dst*8 + c], w[(size_t)e*16 + c] * hs[(size_t)src*8 + c] * s12);
  } else {
    int u = c - 8;
    float base = w[(size_t)e*16 + 8 + u] * hs[(size_t)src*8 + u] * s12;
    atomicAdd(&agg_v[((size_t)0*NNODES + dst)*8 + u], base * sh1[3*e]);
    atomicAdd(&agg_v[((size_t)1*NNODES + dst)*8 + u], base * sh1[3*e+1]);
    atomicAdd(&agg_v[((size_t)2*NNODES + dst)*8 + u], base * sh1[3*e+2]);
  }
}

// Layers 1,2: ns=64, nv=32, ts=96, tv=128, nw=224. hv layout [m][n][32].
__global__ void k_msg(const int* __restrict__ ei, const float* __restrict__ hs,
                      const float* __restrict__ hv, const float* __restrict__ w,
                      const float* __restrict__ sh1,
                      float* __restrict__ agg_s, float* __restrict__ agg_v) {
  long long i = (long long)blockIdx.x * blockDim.x + threadIdx.x;
  if (i >= (long long)NEDGES * 224) return;
  int e = (int)(i / 224), c = (int)(i % 224);
  int src = ei[e], dst = ei[NEDGES + e];
  const float s12 = 0.28867513459481287f;
  float sx = sh1[3*e], sy = sh1[3*e+1], sz = sh1[3*e+2];
  const size_t we = (size_t)e * 224;
  if (c < 64) {                                     // m_s part 1: w_a * hs_e
    atomicAdd(&agg_s[(size_t)dst*96 + c], w[we + c] * hs[(size_t)src*64 + c] * s12);
  } else if (c < 96) {                              // m_s part 2: w_d * dot(hv, sh1)/sqrt(3)
    int u = c - 64;
    float hx = hv[((size_t)0*NNODES + src)*32 + u];
    float hy = hv[((size_t)1*NNODES + src)*32 + u];
    float hz = hv[((size_t)2*NNODES + src)*32 + u];
    float dot = (hx*sx + hy*sy + hz*sz) * 0.5773502691896258f;
    atomicAdd(&agg_s[(size_t)dst*96 + 64 + u], w[we + 160 + u] * dot * s12);
  } else {
    int cv = c - 96;                                // vector channels 0..127
    if (cv < 64) {                                  // (w_b * hs_e) outer sh1
      float base = w[we + 64 + cv] * hs[(size_t)src*64 + cv] * s12;
      atomicAdd(&agg_v[((size_t)0*NNODES + dst)*128 + cv], base * sx);
      atomicAdd(&agg_v[((size_t)1*NNODES + dst)*128 + cv], base * sy);
      atomicAdd(&agg_v[((size_t)2*NNODES + dst)*128 + cv], base * sz);
    } else if (cv < 96) {                           // w_c * hv_e
      int u = cv - 64;
      float wc = w[we + 128 + u] * s12;
      atomicAdd(&agg_v[((size_t)0*NNODES + dst)*128 + 64 + u], wc * hv[((size_t)0*NNODES + src)*32 + u]);
      atomicAdd(&agg_v[((size_t)1*NNODES + dst)*128 + 64 + u], wc * hv[((size_t)1*NNODES + src)*32 + u]);
      atomicAdd(&agg_v[((size_t)2*NNODES + dst)*128 + 64 + u], wc * hv[((size_t)2*NNODES + src)*32 + u]);
    } else {                                        // w_e * cross(hv_e, sh1)/sqrt(2)
      int u = cv - 96;
      float wE = w[we + 192 + u] * s12 * 0.7071067811865475f;
      float hx = hv[((size_t)0*NNODES + src)*32 + u];
      float hy = hv[((size_t)1*NNODES + src)*32 + u];
      float hz = hv[((size_t)2*NNODES + src)*32 + u];
      atomicAdd(&agg_v[((size_t)0*NNODES + dst)*128 + 96 + u], wE * (hy*sz - hz*sy));
      atomicAdd(&agg_v[((size_t)1*NNODES + dst)*128 + 96 + u], wE * (hz*sx - hx*sz));
      atomicAdd(&agg_v[((size_t)2*NNODES + dst)*128 + 96 + u], wE * (hx*sy - hy*sx));
    }
  }
}

// ---------------- skip-combine + nonlinearities ----------------

__global__ void k_combine_s(const float* __restrict__ sc, const float* __restrict__ os,
                            float* __restrict__ xs, float* __restrict__ gate, long long total) {
  long long i = (long long)blockIdx.x * blockDim.x + threadIdx.x;
  if (i >= total) return;                           // total = N*96
  int n = (int)(i / 96), c = (int)(i % 96);
  float comb = C_S * sc[i] + C_X * os[i];
  if (c < 64) xs[(size_t)n*64 + c] = comb * sigm(comb);     // silu
  else        gate[(size_t)n*32 + (c - 64)] = sigm(comb);   // sigmoid gate
}

__global__ void k_combine_v(const float* __restrict__ scv, const float* __restrict__ ov,
                            const float* __restrict__ gate, float* __restrict__ xv,
                            long long total, int has_sc) {
  long long i = (long long)blockIdx.x * blockDim.x + threadIdx.x;
  if (i >= total) return;                           // total = 3N*32, layout [m][n][u]
  int u = (int)(i & 31);
  int n = (int)((i >> 5) % NNODES);
  float comb = has_sc ? (C_S * scv[i] + C_X * ov[i]) : ov[i];
  xv[i] = comb * gate[(size_t)n*32 + u];
}

__global__ void k_final(const float* __restrict__ xv, const float* __restrict__ z,
                        const float* __restrict__ Wout, float* __restrict__ out) {
  int i = blockIdx.x * blockDim.x + threadIdx.x;
  if (i >= NNODES * 3) return;
  int n = i / 3, m = i % 3;
  float acc = 0.f;
#pragma unroll 4
  for (int u = 0; u < 32; ++u) {
    float wz = 0.f;
#pragma unroll
    for (int v = 0; v < 8; ++v) wz += z[n*8 + v] * Wout[u*8 + v];
    acc += xv[((size_t)m*NNODES + n)*32 + u] * wz;
  }
  out[i] = acc * 0.0625f;                           // 1/sqrt(V*Z) = 1/16
}

// ---------------- host orchestration ----------------

static inline int blks(long long n, int t) { return (int)((n + t - 1) / t); }

extern "C" void kernel_launch(void* const* d_in, const int* in_sizes, int n_in,
                              void* d_out, int out_size, void* d_ws, size_t ws_size,
                              hipStream_t stream) {
  (void)in_sizes; (void)n_in; (void)out_size; (void)ws_size;
  const int N = NNODES, E = NEDGES;

  const float* x_s    = (const float*)d_in[0];
  const float* z      = (const float*)d_in[1];
  const int*   ei     = (const int*)d_in[2];
  const float* ev     = (const float*)d_in[3];
  const float* h_edge = (const float*)d_in[4];
  const float* Wout   = (const float*)d_in[39];

  struct LP { const float *lin1_s,*lin1_v,*sc_s,*sc_v,*W1,*b1,*W2,*b2,*W3,*b3,*lin2_s,*lin2_v;
              int ns, nv, ts, tv, nw; };
  LP L[3];
  L[0] = { (const float*)d_in[5], nullptr, (const float*)d_in[6], nullptr,
           (const float*)d_in[7], (const float*)d_in[8], (const float*)d_in[9], (const float*)d_in[10],
           (const float*)d_in[11], (const float*)d_in[12], (const float*)d_in[13], (const float*)d_in[14],
           8, 0, 8, 8, 16 };
  for (int l = 1; l <= 2; ++l) {
    int b = 15 + (l - 1) * 12;
    L[l] = { (const float*)d_in[b+0], (const float*)d_in[b+1], (const float*)d_in[b+2], (const float*)d_in[b+3],
             (const float*)d_in[b+4], (const float*)d_in[b+5], (const float*)d_in[b+6], (const float*)d_in[b+7],
             (const float*)d_in[b+8], (const float*)d_in[b+9], (const float*)d_in[b+10], (const float*)d_in[b+11],
             64, 32, 96, 128, 224 };
  }

  // ---- workspace bump allocator (~460 MB peak) ----
  size_t off = 0;
  auto alloc = [&](size_t bytes) -> void* {
    off = (off + 255) & ~(size_t)255;
    void* p = (char*)d_ws + off; off += bytes; return p;
  };
  float*          sh1      = (float*)alloc((size_t)E*3*4);
  unsigned short* he_bf    = (unsigned short*)alloc((size_t)E*32*2);     // padded K=32
  float*          xs_buf   = (float*)alloc((size_t)N*64*4);
  float*          xv_buf   = (float*)alloc((size_t)3*N*32*4);
  float*          gate     = (float*)alloc((size_t)N*32*4);
  unsigned short* A_s      = (unsigned short*)alloc((size_t)N*768*2);    // also A2_s
  float*          sc_s_buf = (float*)alloc((size_t)N*96*4);
  float*          hs_buf   = (float*)alloc((size_t)N*64*4);
  float*          sc_v_buf = (float*)alloc((size_t)3*N*32*4);
  float*          hv_buf   = (float*)alloc((size_t)3*N*32*4);
  float*          mlp_tmp  = (float*)alloc((size_t)E*64*4);
  unsigned short* t1bf     = (unsigned short*)alloc((size_t)E*32*2);     // padded K=32
  unsigned short* t2bf     = (unsigned short*)alloc((size_t)E*64*2);
  void*           big      = alloc((size_t)E*224*4);     // aliases: A_v (bf16 3N*1024) / wbuf (f32 E*224)
  unsigned short* A_v      = (unsigned short*)big;
  float*          wbuf     = (float*)big;
  float*          agg_s    = (float*)alloc((size_t)N*96*4);
  float*          agg_v    = (float*)alloc((size_t)3*N*128*4);
  float*          out_s_b  = (float*)alloc((size_t)N*96*4);
  float*          out_v_b  = (float*)alloc((size_t)3*N*32*4);
  unsigned short* wt       = (unsigned short*)alloc(256*1024);           // bf16 transposed weight staging

  // stage weight: (Ksrc x Nc) f32 -> (Nc x Kdst) bf16 transposed, zero-padded
  auto stageW = [&](const float* src, int Nc, int Ksrc, int Kdst) {
    long long tot = (long long)Nc * Kdst;
    k_f2bf_T<<<blks(tot, 256), 256, 0, stream>>>(src, wt, tot, Nc, Ksrc, Kdst);
  };
  auto gemm = [&](const unsigned short* A, float* C, int M, int Nc, int K, float scale) {
    long long waves = (long long)(M / 16) * (Nc / 16);
    k_gemm_bf16<<<blks(waves * 32, 256), 256, 0, stream>>>(A, wt, C, M, Nc, K, scale);
  };

  k_prep_sh1<<<blks(E, 256), 256, 0, stream>>>(ev, sh1, E);
  k_cvt_hedge<<<blks((long long)E*32, 256), 256, 0, stream>>>(h_edge, he_bf, (long long)E*32);

  const float* cur_xs = x_s;
  for (int l = 0; l < 3; ++l) {
    const LP& P = L[l];
    int ns = P.ns, nv = P.nv, ts = P.ts, tv = P.tv, nw = P.nw;
    int Ks = ns * 8;                                  // 64 or 512 (multiples of 32)
    float ss = 1.0f / sqrtf((float)Ks);

    // sc_s / lin1_s : A_s = xs (x) z ; GEMMs
    k_build_A<<<blks((long long)N*Ks, 256), 256, 0, stream>>>(cur_xs, z, A_s, (long long)N*Ks, Ks, ns);
    stageW(P.sc_s, 96, Ks, Ks);
    gemm(A_s, sc_s_buf, N, 96, Ks, ss);
    stageW(P.lin1_s, ns, Ks, Ks);
    gemm(A_s, hs_buf, N, ns, Ks, ss);

    if (nv > 0) {                                     // sc_v / lin1_v on stacked 3N rows
      int Kv = nv * 8;                                // 256
      float sv = 1.0f / sqrtf((float)Kv);
      k_build_A<<<blks((long long)3*N*Kv, 256), 256, 0, stream>>>(xv_buf, z, A_v, (long long)3*N*Kv, Kv, nv);
      stageW(P.sc_v, 32, Kv, Kv);
      gemm(A_v, sc_v_buf, 3*N, 32, Kv, sv);
      stageW(P.lin1_v, nv, Kv, Kv);
      gemm(A_v, hv_buf, 3*N, nv, Kv, sv);
    }

    // edge MLP: silu(silu(h W1+b1) W2+b2) W3 + b3   (K=16 padded to 32)
    stageW(P.W1, 16, 16, 32);
    gemm(he_bf, mlp_tmp, E, 16, 32, 1.0f);
    k_bias_silu_bf<<<blks((long long)E*32, 256), 256, 0, stream>>>(mlp_tmp, P.b1, t1bf,
                                                                   (long long)E*32, 16, 32);
    stageW(P.W2, 64, 16, 32);
    gemm(t1bf, mlp_tmp, E, 64, 32, 1.0f);
    k_bias_silu_bf<<<blks((long long)E*64, 256), 256, 0, stream>>>(mlp_tmp, P.b2, t2bf,
                                                                   (long long)E*64, 64, 64);
    stageW(P.W3, nw, 64, 64);
    gemm(t2bf, wbuf, E, nw, 64, 1.0f);
    k_bias_inplace<<<blks((long long)E*nw, 256), 256, 0, stream>>>(wbuf, P.b3, (long long)E*nw, nw);

    // scatter-sum messages
    hipMemsetAsync(agg_s, 0, (size_t)N * ts * 4, stream);
    hipMemsetAsync(agg_v, 0, (size_t)3 * N * tv * 4, stream);
    if (l == 0)
      k_msg0<<<blks((long long)E*16, 256), 256, 0, stream>>>(ei, hs_buf, wbuf, sh1, agg_s, agg_v);
    else
      k_msg<<<blks((long long)E*224, 256), 256, 0, stream>>>(ei, hs_buf, hv_buf, wbuf, sh1, agg_s, agg_v);

    // lin2 (wbuf dead -> BIG region reused for A2_v)
    int K2s = ts * 8, K2v = tv * 8;                   // 64/768, 64/1024 (multiples of 32)
    k_build_A<<<blks((long long)N*K2s, 256), 256, 0, stream>>>(agg_s, z, A_s, (long long)N*K2s, K2s, ts);
    stageW(P.lin2_s, 96, K2s, K2s);
    gemm(A_s, out_s_b, N, 96, K2s, 1.0f / sqrtf((float)K2s));
    k_build_A<<<blks((long long)3*N*K2v, 256), 256, 0, stream>>>(agg_v, z, A_v, (long long)3*N*K2v, K2v, tv);
    stageW(P.lin2_v, 32, K2v, K2v);
    gemm(A_v, out_v_b, 3*N, 32, K2v, 1.0f / sqrtf((float)K2v));

    // skip + nonlinearity
    k_combine_s<<<blks((long long)N*96, 256), 256, 0, stream>>>(sc_s_buf, out_s_b, xs_buf, gate, (long long)N*96);
    k_combine_v<<<blks((long long)3*N*32, 256), 256, 0, stream>>>(sc_v_buf, out_v_b, gate, xv_buf,
                                                                  (long long)3*N*32, l > 0 ? 1 : 0);
    cur_xs = xs_buf;
  }

  k_final<<<blks((long long)N*3, 256), 256, 0, stream>>>(xv_buf, z, Wout, (float*)d_out);
}
